// VectorQuantizer_63668595196458
// MI455X (gfx1250) — matrix operand order repaired and checked by reading, not compiled, not run on hardware
//
#include <hip/hip_runtime.h>
#include <hip/hip_bf16.h>

// ---------------------------------------------------------------------------
// VQ-VAE vector quantizer forward, MI455X (gfx1250, wave32).
// Distance GEMM done with v_wmma_f32_16x16x32_bf16 using split-bf16
// (hi + lo residual) for ~fp32 accuracy: f.e ~= fh.eh + fh.el + fl.eh.
// ---------------------------------------------------------------------------

typedef __attribute__((ext_vector_type(16))) __bf16 v16bf;
typedef __attribute__((ext_vector_type(8)))  float  v8f;

#define KCB   8192      // codebook entries
#define DDIM  512       // embedding dim
#define NTOK  32768     // B*T
#define BETA  0.25f

#define WGSIZE     256  // 8 wave32s
#define BLOCK_ROWS 128  // N-rows per workgroup (16 per wave)
#define KBLK       128  // codebook columns per accumulation block
#define DCHUNK     32   // D elements per WMMA step

// ---------------------------------------------------------------------------
// Kernel 1: ||e_k||^2 per codebook row (one wave per row); zero correct ctr.
// ---------------------------------------------------------------------------
__global__ void vq_prep(const float* __restrict__ emb,
                        float* __restrict__ enorm,
                        int* __restrict__ counter) {
    if (blockIdx.x == 0 && threadIdx.x == 0) *counter = 0;
    int gid  = blockIdx.x * blockDim.x + threadIdx.x;
    int row  = gid >> 5;
    int lane = gid & 31;
    if (row >= KCB) return;
    const float* e = emb + (size_t)row * DDIM;
    float s = 0.f;
    #pragma unroll
    for (int d = lane * 4; d < DDIM; d += 128) {
        float4 v = *(const float4*)(e + d);
        s += v.x * v.x + v.y * v.y + v.z * v.z + v.w * v.w;
    }
    #pragma unroll
    for (int m = 16; m; m >>= 1) s += __shfl_xor(s, m, 32);
    if (lane == 0) enorm[row] = s;
}

// ---------------------------------------------------------------------------
// Kernel 2: fused  argmin_k( ||e_k||^2 - 2 f.e_k )  via split-bf16 WMMA.
// ---------------------------------------------------------------------------
__global__ __launch_bounds__(WGSIZE)
void vq_argmin(const float* __restrict__ lat,
               const float* __restrict__ emb,
               const float* __restrict__ enorm,
               float* __restrict__ inds_out) {
    // B block staged as split bf16, [col][d] layout, 8KB + 8KB.
    __shared__ __align__(32) __bf16 sBh[KBLK * DCHUNK];
    __shared__ __align__(32) __bf16 sBl[KBLK * DCHUNK];

    const int t      = threadIdx.x;
    const int lane   = t & 31;
    const int waveId = t >> 5;
    const int m      = lane & 15;   // A row within tile / B,C column within tile
    const int h      = lane >> 4;   // lane half
    const int rowBase = blockIdx.x * BLOCK_ROWS + waveId * 16;

    const v8f vzero = {0.f, 0.f, 0.f, 0.f, 0.f, 0.f, 0.f, 0.f};

    float minv[8];
    int   mini[8];
    #pragma unroll
    for (int r = 0; r < 8; ++r) { minv[r] = __builtin_inff(); mini[r] = 0x7fffffff; }

    for (int col0 = 0; col0 < KCB; col0 += KBLK) {
        v8f acc[8];
        #pragma unroll
        for (int c = 0; c < 8; ++c) acc[c] = vzero;

        for (int d0 = 0; d0 < DDIM; d0 += DCHUNK) {
            __syncthreads();   // previous block's LDS reads done
            // Cooperative load of B block (KBLK cols x DCHUNK d), split to
            // bf16 hi + residual lo. 256 threads x 16 elems = 128x32.
            {
                const float* gB = emb + (size_t)col0 * DDIM + d0;
                #pragma unroll
                for (int i = 0; i < 16; i += 4) {
                    int e   = t * 16 + i;
                    int col = e >> 5;
                    int d   = e & 31;
                    float4 f = *(const float4*)(gB + (size_t)col * DDIM + d);
                    __bf16 h0 = (__bf16)f.x, h1 = (__bf16)f.y;
                    __bf16 h2 = (__bf16)f.z, h3 = (__bf16)f.w;
                    int o = col * DCHUNK + d;
                    sBh[o] = h0; sBh[o + 1] = h1; sBh[o + 2] = h2; sBh[o + 3] = h3;
                    sBl[o]     = (__bf16)(f.x - (float)h0);
                    sBl[o + 1] = (__bf16)(f.y - (float)h1);
                    sBl[o + 2] = (__bf16)(f.z - (float)h2);
                    sBl[o + 3] = (__bf16)(f.w - (float)h3);
                }
            }
            __syncthreads();

            // A 16x32 tile -> split-bf16 regs, contiguous K layout:
            // vgpr j, lane half h holds K = 16h + 2j and +1.
            v16bf ah, al;
            {
                const float* gA = lat + (size_t)(rowBase + m) * DDIM + d0;
                #pragma unroll
                for (int j = 0; j < 8; ++j) {
                    int pk = 16 * h + 2 * j;
                    float2 f = *(const float2*)(gA + pk);
                    __bf16 x0 = (__bf16)f.x, x1 = (__bf16)f.y;
                    ah[2 * j]     = x0;
                    ah[2 * j + 1] = x1;
                    al[2 * j]     = (__bf16)(f.x - (float)x0);
                    al[2 * j + 1] = (__bf16)(f.y - (float)x1);
                }
            }

            // 8 column tiles x 3 split-product WMMAs.
            #pragma unroll
            for (int c = 0; c < 8; ++c) {
                // B layout: lane half h supplies K = 16h..16h+15 for column m.
                const v16bf bh = *(const v16bf*)&sBh[(c * 16 + m) * DCHUNK + h * 16];
                const v16bf bl = *(const v16bf*)&sBl[(c * 16 + m) * DCHUNK + h * 16];
                acc[c] = __builtin_amdgcn_wmma_f32_16x16x32_bf16(
                    false, ah, false, bh, (short)0, acc[c], false, false);
                acc[c] = __builtin_amdgcn_wmma_f32_16x16x32_bf16(
                    false, ah, false, bl, (short)0, acc[c], false, false);
                acc[c] = __builtin_amdgcn_wmma_f32_16x16x32_bf16(
                    false, al, false, bh, (short)0, acc[c], false, false);
            }
        }

        // Epilogue: dist = ||e||^2 - 2 f.e ; running per-lane min/argmin.
        // C layout: lane (m,h), vgpr r holds row (8h + r), column m of tile.
        #pragma unroll
        for (int c = 0; c < 8; ++c) {
            int   kidx = col0 + c * 16 + m;
            float en   = enorm[kidx];
            #pragma unroll
            for (int r = 0; r < 8; ++r) {
                float dist = en - 2.0f * acc[c][r];
                if (dist < minv[r]) { minv[r] = dist; mini[r] = kidx; }
            }
        }
    }

    // Reduce across the 16 lanes of each half (columns), smaller-index ties.
    #pragma unroll
    for (int r = 0; r < 8; ++r) {
        float v = minv[r];
        int   i = mini[r];
        #pragma unroll
        for (int msk = 1; msk <= 8; msk <<= 1) {
            float ov = __shfl_xor(v, msk, 32);
            int   oi = __shfl_xor(i, msk, 32);
            if (ov < v || (ov == v && oi < i)) { v = ov; i = oi; }
        }
        if (m == 0) inds_out[rowBase + h * 8 + r] = (float)i;  // exact in f32
    }
}

// ---------------------------------------------------------------------------
// Kernel 3: gather quantized rows, vq_loss = (1+beta)*mse, correct count.
// One wave per token row.
// ---------------------------------------------------------------------------
__global__ void vq_gather(const float* __restrict__ lat,
                          const float* __restrict__ emb,
                          const float* __restrict__ indsF,
                          const int* __restrict__ gold,
                          float* __restrict__ q_out,
                          float* __restrict__ vq_out,
                          int* __restrict__ counter) {
    int gid  = blockIdx.x * blockDim.x + threadIdx.x;
    int row  = gid >> 5;
    int lane = gid & 31;
    if (row >= NTOK) return;
    int idx = (int)indsF[row];
    const float* e = emb + (size_t)idx * DDIM;
    const float* f = lat + (size_t)row * DDIM;
    float*       q = q_out + (size_t)row * DDIM;
    float s = 0.f;
    #pragma unroll
    for (int d = lane * 4; d < DDIM; d += 128) {
        float4 ev = *(const float4*)(e + d);
        float4 fv = *(const float4*)(f + d);
        *(float4*)(q + d) = ev;   // straight-through forward == quantized
        float dx = ev.x - fv.x, dy = ev.y - fv.y;
        float dz = ev.z - fv.z, dw = ev.w - fv.w;
        s += dx * dx + dy * dy + dz * dz + dw * dw;
    }
    #pragma unroll
    for (int msk = 16; msk; msk >>= 1) s += __shfl_xor(s, msk, 32);
    if (lane == 0) {
        vq_out[row] = (1.0f + BETA) * (s * (1.0f / (float)DDIM));
        if (gold[row] == idx) atomicAdd(counter, 1);
    }
}

// ---------------------------------------------------------------------------
// Kernel 4: finalize (correct, total).
// ---------------------------------------------------------------------------
__global__ void vq_final(const int* __restrict__ counter, float* __restrict__ stats) {
    stats[0] = (float)(*counter);
    stats[1] = (float)NTOK;
}

// ---------------------------------------------------------------------------
extern "C" void kernel_launch(void* const* d_in, const int* in_sizes, int n_in,
                              void* d_out, int out_size, void* d_ws, size_t ws_size,
                              hipStream_t stream) {
    (void)in_sizes; (void)n_in; (void)out_size; (void)ws_size;
    // setup_inputs order: gold_encoding_inds(int32), latents(f32), epc(int), emb(f32)
    const int*   gold = (const int*)d_in[0];
    const float* lat  = (const float*)d_in[1];
    const float* emb  = (const float*)d_in[3];

    float* out      = (float*)d_out;
    float* q_out    = out;                               // [N, D]
    float* vq_out   = out + (size_t)NTOK * DDIM;         // [N]
    float* inds_out = vq_out + NTOK;                     // [N] (as float)
    float* stats    = inds_out + NTOK;                   // [correct, total]

    float* enorm   = (float*)d_ws;                       // K floats
    int*   counter = (int*)((char*)d_ws + (size_t)KCB * sizeof(float));

    vq_prep  <<<KCB / 8,            256,    0, stream>>>(emb, enorm, counter);
    vq_argmin<<<NTOK / BLOCK_ROWS,  WGSIZE, 0, stream>>>(lat, emb, enorm, inds_out);
    vq_gather<<<NTOK / 8,           256,    0, stream>>>(lat, emb, inds_out, gold,
                                                         q_out, vq_out, counter);
    vq_final <<<1, 1, 0, stream>>>(counter, stats);
}
